// SWINCompass_74199855005908
// MI455X (gfx1250) — matrix operand order repaired and checked by reading, not compiled, not run on hardware
//
#include <hip/hip_runtime.h>
#include <hip/hip_bf16.h>
#include <math.h>

// ---------------------------------------------------------------------------
// Video Swin-T forward for MI455X (gfx1250, wave32, WMMA 16x16x32 f16->f32)
// ---------------------------------------------------------------------------

typedef __attribute__((ext_vector_type(16))) _Float16 v16h;
typedef __attribute__((ext_vector_type(8)))  _Float16 v8h;
typedef __attribute__((ext_vector_type(8)))  float    v8f;
typedef __attribute__((ext_vector_type(4)))  float    v4f;

#define WAVE 32

// ---------------------------------------------------------------------------
// Patch embedding: 4x4x4/stride-4 conv, 1 -> 48 channels, output (b,d,h,w,c)
// All addresses 16B aligned -> float4 loads.
// ---------------------------------------------------------------------------
__global__ __launch_bounds__(256)
void patch_embed_kernel(const float* __restrict__ x,
                        const float* __restrict__ w,
                        const float* __restrict__ bias,
                        float* __restrict__ out,
                        int B_, int Din, int Do, int Cout) {
  size_t idx = (size_t)blockIdx.x * blockDim.x + threadIdx.x;
  size_t total = (size_t)B_ * Do * Do * Do * Cout;
  if (idx >= total) return;
  int cc = idx % Cout;
  size_t t = idx / Cout;
  int wv = t % Do; t /= Do;
  int hv = t % Do; t /= Do;
  int dv = t % Do;
  int b  = (int)(t / Do);
  const float* xp = x + (((size_t)b * Din + dv * 4) * Din + hv * 4) * Din + wv * 4;
  const float* wp = w + (size_t)cc * 64;
  float acc = bias[cc];
  #pragma unroll
  for (int i = 0; i < 4; ++i)
    #pragma unroll
    for (int j = 0; j < 4; ++j) {
      v4f xr = *reinterpret_cast<const v4f*>(xp + ((size_t)i * Din + j) * Din);
      v4f wr = *reinterpret_cast<const v4f*>(wp + (i * 4 + j) * 4);
      acc += xr.x * wr.x + xr.y * wr.y + xr.z * wr.z + xr.w * wr.w;
    }
  out[idx] = acc;
}

// ---------------------------------------------------------------------------
// LayerNorm: one wave per row, shuffle reductions (wave32).
// ---------------------------------------------------------------------------
__global__ __launch_bounds__(WAVE)
void layernorm_kernel(const float* __restrict__ in,
                      const float* __restrict__ g,
                      const float* __restrict__ b,
                      float* __restrict__ out,
                      int rows, int C) {
  int row = blockIdx.x;
  if (row >= rows) return;
  int lane = threadIdx.x;
  const float* x = in + (size_t)row * C;
  float s = 0.f;
  for (int i = lane; i < C; i += WAVE) s += x[i];
  #pragma unroll
  for (int off = 16; off >= 1; off >>= 1) s += __shfl_xor(s, off, WAVE);
  float mu = s / (float)C;
  float v = 0.f;
  for (int i = lane; i < C; i += WAVE) { float d = x[i] - mu; v += d * d; }
  #pragma unroll
  for (int off = 16; off >= 1; off >>= 1) v += __shfl_xor(v, off, WAVE);
  float inv = rsqrtf(v / (float)C + 1e-5f);
  float* o = out + (size_t)row * C;
  if (g != nullptr) {
    for (int i = lane; i < C; i += WAVE) o[i] = (x[i] - mu) * inv * g[i] + b[i];
  } else {
    for (int i = lane; i < C; i += WAVE) o[i] = (x[i] - mu) * inv;
  }
}

// ---------------------------------------------------------------------------
// GEMM: C[M,N] = act(A[M,K] * W[N,K]^T + bias), K % 16 == 0 guaranteed.
// 128-thread block = 4 waves computing a 32x32 tile (2x A / 2x B LDS reuse).
// Branchless row guards (clamped addresses + select); float4 global loads;
// 16B-aligned v8h LDS stores.  act: 0 = none, 1 = exact GELU.
// ---------------------------------------------------------------------------
__global__ __launch_bounds__(128)
void gemm_wmma_kernel(const float* __restrict__ A,
                      const float* __restrict__ W,
                      const float* __restrict__ bias,
                      float* __restrict__ C,
                      int M, int N, int K, int act) {
  // row stride 40 halves = 80 B (multiple of 16 B)
  __shared__ _Float16 As[32][40];
  __shared__ _Float16 Bs[32][40];
  const int tid  = threadIdx.x;
  const int lane = tid & 31;
  const int wave = tid >> 5;          // 0..3
  const int r    = lane & 15;
  const int half = lane >> 4;
  const int wm   = wave >> 1;         // 0..1 : row quadrant
  const int wn   = wave & 1;          // 0..1 : col quadrant
  const int m0 = blockIdx.y * 32;
  const int n0 = blockIdx.x * 32;
  // staging: thread covers (row = lane, cols chunk*8..chunk*8+7), chunk = wave
  const int srow = lane;
  const int schunk = wave;

  const int arow  = m0 + srow;
  const int wrow  = n0 + srow;
  const int arow_c = (arow < M) ? arow : (M - 1);
  const int wrow_c = (wrow < N) ? wrow : (N - 1);
  const float avalid = (arow < M) ? 1.f : 0.f;
  const float wvalid = (wrow < N) ? 1.f : 0.f;

  v8f acc = {};
  for (int k0 = 0; k0 < K; k0 += 32) {
    const int kc = k0 + schunk * 8;   // uniform per wave
    v8h ha = {}, hb = {};
    if (kc < K) {                     // K%8==0 -> kc<K implies kc+8<=K
      const v4f* ap = reinterpret_cast<const v4f*>(A + (size_t)arow_c * K + kc);
      v4f f0 = ap[0], f1 = ap[1];
      const v4f* wp = reinterpret_cast<const v4f*>(W + (size_t)wrow_c * K + kc);
      v4f g0 = wp[0], g1 = wp[1];
      ha[0] = (_Float16)(f0.x * avalid); ha[1] = (_Float16)(f0.y * avalid);
      ha[2] = (_Float16)(f0.z * avalid); ha[3] = (_Float16)(f0.w * avalid);
      ha[4] = (_Float16)(f1.x * avalid); ha[5] = (_Float16)(f1.y * avalid);
      ha[6] = (_Float16)(f1.z * avalid); ha[7] = (_Float16)(f1.w * avalid);
      hb[0] = (_Float16)(g0.x * wvalid); hb[1] = (_Float16)(g0.y * wvalid);
      hb[2] = (_Float16)(g0.z * wvalid); hb[3] = (_Float16)(g0.w * wvalid);
      hb[4] = (_Float16)(g1.x * wvalid); hb[5] = (_Float16)(g1.y * wvalid);
      hb[6] = (_Float16)(g1.z * wvalid); hb[7] = (_Float16)(g1.w * wvalid);
    }
    *reinterpret_cast<v8h*>(&As[srow][schunk * 8]) = ha;
    *reinterpret_cast<v8h*>(&Bs[srow][schunk * 8]) = hb;
    __syncthreads();

    // A fragment: row = wm*16 + r ; halves 0..7 -> K base half*8,
    // halves 8..15 -> K base half*8+16  (CDNA5 16-bit A 16x32 layout)
    const int ar = wm * 16 + r;
    v8h alo = *reinterpret_cast<const v8h*>(&As[ar][half * 8]);
    v8h ahi = *reinterpret_cast<const v8h*>(&As[ar][half * 8 + 16]);
    v16h a = __builtin_shufflevector(alo, ahi, 0, 1, 2, 3, 4, 5, 6, 7,
                                     8, 9, 10, 11, 12, 13, 14, 15);
    // B fragment: column = wn*16 + r ; same K layout as A
    const int br = wn * 16 + r;
    v8h blo = *reinterpret_cast<const v8h*>(&Bs[br][half * 8]);
    v8h bhi = *reinterpret_cast<const v8h*>(&Bs[br][half * 8 + 16]);
    v16h b = __builtin_shufflevector(blo, bhi, 0, 1, 2, 3, 4, 5, 6, 7,
                                     8, 9, 10, 11, 12, 13, 14, 15);
    acc = __builtin_amdgcn_wmma_f32_16x16x32_f16(false, a, false, b,
                                                 (short)0, acc, false, false);
    __syncthreads();
  }
  const int n = n0 + wn * 16 + r;
  float bv = (bias != nullptr && n < N) ? bias[n] : 0.f;
  #pragma unroll
  for (int rr = 0; rr < 8; ++rr) {
    int m = m0 + wm * 16 + half * 8 + rr;
    if (m < M && n < N) {
      float v = acc[rr] + bv;
      if (act == 1) v = 0.5f * v * (1.f + erff(v * 0.70710678118654752f));
      C[(size_t)m * N + n] = v;
    }
  }
}

// ---------------------------------------------------------------------------
// Window partition with pad + negative roll.
// ---------------------------------------------------------------------------
__global__ __launch_bounds__(256)
void win_gather_kernel(const float* __restrict__ xn,
                       float* __restrict__ xw,
                       int D, int H, int W, int Cc,
                       int wsd, int wsh, int wsw,
                       int nwd, int nwh, int nww,
                       int dp, int hp, int wpad, int ss,
                       size_t total) {
  size_t idx = (size_t)blockIdx.x * blockDim.x + threadIdx.x;
  if (idx >= total) return;
  int c = idx % Cc;
  size_t tok = idx / Cc;
  int n = wsd * wsh * wsw;
  int pos = tok % n;
  int win = (int)(tok / n);
  int nwin = nwd * nwh * nww;
  int b = win / nwin;
  int wl = win % nwin;
  int wd = wl / (nwh * nww), wh = (wl / nww) % nwh, ww = wl % nww;
  int pd = pos / (wsh * wsw), ph = (pos / wsw) % wsh, pw = pos % wsw;
  int gd = wd * wsd + pd, gh = wh * wsh + ph, gw = ww * wsw + pw;
  int sd = (gd + ss) % dp, sh = (gh + ss) % hp, sw = (gw + ss) % wpad;
  float v = 0.f;
  if (sd < D && sh < H && sw < W)
    v = xn[((((size_t)b * D + sd) * H + sh) * W + sw) * Cc + c];
  xw[idx] = v;
}

// ---------------------------------------------------------------------------
// Window reverse + positive roll + crop + residual add.
// ---------------------------------------------------------------------------
__global__ __launch_bounds__(256)
void win_scatter_res_kernel(const float* __restrict__ x,
                            const float* __restrict__ win,
                            float* __restrict__ out,
                            int D, int H, int W, int Cc,
                            int wsd, int wsh, int wsw,
                            int nwd, int nwh, int nww,
                            int dp, int hp, int wpad, int ss,
                            size_t total) {
  size_t idx = (size_t)blockIdx.x * blockDim.x + threadIdx.x;
  if (idx >= total) return;
  int c = idx % Cc;
  size_t t = idx / Cc;
  int w = t % W; t /= W;
  int h = t % H; t /= H;
  int d = t % D;
  int b = (int)(t / D);
  int gd = (d - ss + dp) % dp, gh = (h - ss + hp) % hp, gw = (w - ss + wpad) % wpad;
  int wd = gd / wsd, pd = gd % wsd;
  int wh = gh / wsh, ph = gh % wsh;
  int ww = gw / wsw, pw = gw % wsw;
  int n = wsd * wsh * wsw;
  int wi = ((b * nwd + wd) * nwh + wh) * nww + ww;
  int pos = (pd * wsh + ph) * wsw + pw;
  out[idx] = x[idx] + win[((size_t)wi * n + pos) * Cc + c];
}

// ---------------------------------------------------------------------------
// Flash-style shifted-window attention (head_dim == 16 for every stage).
// One wave = one (window, head, 16-query block). Streams 16-key blocks,
// QK^T and P*V each one v_wmma_f32_16x16x32_f16 (upper K half zero-padded).
// Branchless clamped-address loads, float4-vectorized; online softmax with
// 16-lane butterfly reductions.  Relative-position bias uses the reference's
// fixed 7^3 index table (sliced), region mask computed analytically.
// ---------------------------------------------------------------------------
__global__ __launch_bounds__(WAVE)
void attn_flash_kernel(const float* __restrict__ qkv,
                       float* __restrict__ out,
                       const float* __restrict__ rpb,
                       int n, int c, int nh,
                       int wsd, int wsh, int wsw,
                       int nwd, int nwh, int nww,
                       int dp, int hp, int wpad,
                       int ss, int shifted) {
  // row strides 24 halves = 48 B (multiple of 16 B)
  __shared__ _Float16 Ps[16][24];
  __shared__ _Float16 Vs[16][24];
  const int lane = threadIdx.x;
  const int r = lane & 15;
  const int half = lane >> 4;
  const int win = blockIdx.z;
  const int head = blockIdx.y;
  const int i0 = blockIdx.x * 16;
  const int nwin = nwd * nwh * nww;
  const int wl = win % nwin;
  const int wd = wl / (nwh * nww), wh = (wl / nww) % nwh, ww = wl % nww;
  const size_t c3 = (size_t)3 * c;
  const float scale = 0.25f;  // hd=16 -> hd^-0.5

  // ---- Q fragment (constant over key blocks); K = 0..15 live, rest zero ----
  v16h aq = {};
  {
    int i = i0 + r;
    float qv = (i < n) ? scale : 0.f;
    int ic = (i < n) ? i : (n - 1);
    const v4f* qp = reinterpret_cast<const v4f*>(
        qkv + ((size_t)win * n + ic) * c3 + (size_t)head * 16 + half * 8);
    v4f q0 = qp[0], q1 = qp[1];
    aq[0] = (_Float16)(q0.x * qv); aq[1] = (_Float16)(q0.y * qv);
    aq[2] = (_Float16)(q0.z * qv); aq[3] = (_Float16)(q0.w * qv);
    aq[4] = (_Float16)(q1.x * qv); aq[5] = (_Float16)(q1.y * qv);
    aq[6] = (_Float16)(q1.z * qv); aq[7] = (_Float16)(q1.w * qv);
  }
  // per-lane query-row metadata (8 rows in C/D layout)
  int q7d[8], q7h[8], q7w[8], qreg[8];
  #pragma unroll
  for (int rr = 0; rr < 8; ++rr) {
    int i = i0 + half * 8 + rr;
    int ii = (i < n) ? i : 0;
    q7d[rr] = ii / 49; q7h[rr] = (ii / 7) % 7; q7w[rr] = ii % 7;
    int pd = ii / (wsh * wsw), ph = (ii / wsw) % wsh, pw = ii % wsw;
    int td = wd * wsd + pd, th = wh * wsh + ph, tw = ww * wsw + pw;
    int rd = (td < dp - wsd) ? 0 : ((td < dp - ss) ? 1 : 2);
    int rh = (th < hp - wsh) ? 0 : ((th < hp - ss) ? 1 : 2);
    int rw = (tw < wpad - wsw) ? 0 : ((tw < wpad - ss) ? 1 : 2);
    qreg[rr] = rd * 9 + rh * 3 + rw;
  }

  float m_[8], l_[8];
  v8f acco = {};
  #pragma unroll
  for (int rr = 0; rr < 8; ++rr) { m_[rr] = -1e30f; l_[rr] = 0.f; }

  for (int j0 = 0; j0 < n; j0 += 16) {
    const int j = j0 + r;
    const int kvalid = (j < n);
    const int jj = kvalid ? j : (n - 1);
    const float kval = kvalid ? 1.f : 0.f;
    const size_t krow = ((size_t)win * n + jj) * c3 + (size_t)head * 16;

    // K fragment: lanes 0..15 hold key column j, contraction K = 0..15.
    v16h bk = {};
    {
      const v4f* kp = reinterpret_cast<const v4f*>(qkv + krow + c + half * 8);
      v4f k0 = kp[0], k1 = kp[1];
      bk[0]  = (_Float16)(k0.x * kval); bk[1]  = (_Float16)(k0.y * kval);
      bk[2]  = (_Float16)(k0.z * kval); bk[3]  = (_Float16)(k0.w * kval);
      bk[4]  = (_Float16)(k1.x * kval); bk[5]  = (_Float16)(k1.y * kval);
      bk[6]  = (_Float16)(k1.z * kval); bk[7]  = (_Float16)(k1.w * kval);
    }
    v8f S = {};
    S = __builtin_amdgcn_wmma_f32_16x16x32_f16(false, aq, false, bk,
                                               (short)0, S, false, false);
    // key-column metadata
    int k7d = jj / 49, k7h = (jj / 7) % 7, k7w = jj % 7;
    int kreg = 0;
    if (shifted) {
      int pd = jj / (wsh * wsw), ph = (jj / wsw) % wsh, pw = jj % wsw;
      int td = wd * wsd + pd, th = wh * wsh + ph, tw = ww * wsw + pw;
      int rd = (td < dp - wsd) ? 0 : ((td < dp - ss) ? 1 : 2);
      int rh = (th < hp - wsh) ? 0 : ((th < hp - ss) ? 1 : 2);
      int rw = (tw < wpad - wsw) ? 0 : ((tw < wpad - ss) ? 1 : 2);
      kreg = rd * 9 + rh * 3 + rw;
    }
    // stage V block (16 keys x 16 dims) into LDS, vectorized
    {
      const v4f* vp = reinterpret_cast<const v4f*>(qkv + krow + 2 * (size_t)c + half * 8);
      v4f u0 = vp[0], u1 = vp[1];
      v8h hv;
      hv[0] = (_Float16)(u0.x * kval); hv[1] = (_Float16)(u0.y * kval);
      hv[2] = (_Float16)(u0.z * kval); hv[3] = (_Float16)(u0.w * kval);
      hv[4] = (_Float16)(u1.x * kval); hv[5] = (_Float16)(u1.y * kval);
      hv[6] = (_Float16)(u1.z * kval); hv[7] = (_Float16)(u1.w * kval);
      *reinterpret_cast<v8h*>(&Vs[r][half * 8]) = hv;
    }
    // bias + mask, online softmax update
    #pragma unroll
    for (int rr = 0; rr < 8; ++rr) {
      int i = i0 + half * 8 + rr;
      float s = -1e30f;
      if (i < n && kvalid) {
        int rel = (q7d[rr] - k7d + 6) * 169 + (q7h[rr] - k7h + 6) * 13 +
                  (q7w[rr] - k7w + 6);
        s = S[rr] + rpb[(size_t)rel * nh + head];
        if (shifted && qreg[rr] != kreg) s -= 100.f;
      }
      float mx = s;
      #pragma unroll
      for (int off = 1; off < 16; off <<= 1)
        mx = fmaxf(mx, __shfl_xor(mx, off, WAVE));
      float mnew = fmaxf(m_[rr], mx);
      float alpha = __expf(m_[rr] - mnew);
      float p = __expf(s - mnew);
      float rs = p;
      #pragma unroll
      for (int off = 1; off < 16; off <<= 1)
        rs += __shfl_xor(rs, off, WAVE);
      l_[rr] = l_[rr] * alpha + rs;
      m_[rr] = mnew;
      acco[rr] *= alpha;
      Ps[half * 8 + rr][r] = (_Float16)p;
    }
    __syncthreads();
    // P fragment (A layout): row = r, contraction key = half*8 + h (<16)
    v16h ap = {}, bv = {};
    {
      v8h plo = *reinterpret_cast<const v8h*>(&Ps[r][half * 8]);
      #pragma unroll
      for (int h2 = 0; h2 < 8; ++h2) ap[h2] = plo[h2];
    }
    // V fragment (B layout): column = hd dim r, contraction key = half*8 + h (<16)
    {
      #pragma unroll
      for (int h2 = 0; h2 < 8; ++h2) bv[h2] = Vs[half * 8 + h2][r];
    }
    acco = __builtin_amdgcn_wmma_f32_16x16x32_f16(false, ap, false, bv,
                                                  (short)0, acco, false, false);
    __syncthreads();
  }
  #pragma unroll
  for (int rr = 0; rr < 8; ++rr) {
    int i = i0 + half * 8 + rr;
    if (i < n) {
      float denom = fmaxf(l_[rr], 1e-20f);
      out[((size_t)win * n + i) * c + (size_t)head * 16 + r] = acco[rr] / denom;
    }
  }
}

// ---------------------------------------------------------------------------
// Elementwise residual add.
// ---------------------------------------------------------------------------
__global__ __launch_bounds__(256)
void add_kernel(const float* __restrict__ a,
                const float* __restrict__ b,
                float* __restrict__ o, size_t total) {
  size_t idx = (size_t)blockIdx.x * blockDim.x + threadIdx.x;
  if (idx < total) o[idx] = a[idx] + b[idx];
}

// ---------------------------------------------------------------------------
// Patch-merging gather: (b,D,H,W,C) -> (b,D2,H2,W2,8C), zero-padded odd dims.
// ---------------------------------------------------------------------------
__global__ __launch_bounds__(256)
void merge_gather_kernel(const float* __restrict__ x,
                         float* __restrict__ out,
                         int D, int H, int W, int Cc,
                         int D2, int H2, int W2, size_t total) {
  size_t idx = (size_t)blockIdx.x * blockDim.x + threadIdx.x;
  if (idx >= total) return;
  int C8 = 8 * Cc;
  int c8 = idx % C8;
  size_t t = idx / C8;
  int wv = t % W2; t /= W2;
  int hv = t % H2; t /= H2;
  int dv = t % D2;
  int b = (int)(t / D2);
  int part = c8 / Cc, c = c8 % Cc;
  const int po[8][3] = {{0,0,0},{1,0,0},{0,1,0},{0,0,1},
                        {1,1,0},{1,0,1},{0,1,1},{1,1,1}};
  int d = 2 * dv + po[part][0];
  int h = 2 * hv + po[part][1];
  int w = 2 * wv + po[part][2];
  float v = 0.f;
  if (d < D && h < H && w < W)
    v = x[((((size_t)b * D + d) * H + h) * W + w) * Cc + c];
  out[idx] = v;
}

// ---------------------------------------------------------------------------
// Host-side orchestration
// ---------------------------------------------------------------------------
static inline unsigned cdiv(size_t a, unsigned b) { return (unsigned)((a + b - 1) / b); }

struct BlockP {
  const float *fc1_b, *fc1_w, *fc2_b, *fc2_w;
  const float *n1b, *n1g, *n2b, *n2g;
  const float *pb, *pw, *qb, *qw, *rpb;
};

extern "C" void kernel_launch(void* const* d_in, const int* in_sizes, int n_in,
                              void* d_out, int out_size, void* d_ws, size_t ws_size,
                              hipStream_t stream) {
  (void)in_sizes; (void)n_in; (void)out_size; (void)ws_size;
  const int B = 2;
  const int NUM_HEADS_H[4] = {3, 6, 12, 24};

  // Input order: x first, then params flattened in JAX sorted-dict-key order.
  const float* xin = (const float*)d_in[0];
  int p = 1;
  const float* cls_b = (const float*)d_in[p++];
  const float* cls_w = (const float*)d_in[p++];
  BlockP blk[4][2];
  const float *dn_b[4], *dn_g[4], *red[4];
  for (int L = 0; L < 4; ++L) {
    for (int j = 0; j < 2; ++j) {
      BlockP& bp = blk[L][j];
      bp.fc1_b = (const float*)d_in[p++]; bp.fc1_w = (const float*)d_in[p++];
      bp.fc2_b = (const float*)d_in[p++]; bp.fc2_w = (const float*)d_in[p++];
      bp.n1b = (const float*)d_in[p++];   bp.n1g = (const float*)d_in[p++];
      bp.n2b = (const float*)d_in[p++];   bp.n2g = (const float*)d_in[p++];
      bp.pb = (const float*)d_in[p++];    bp.pw = (const float*)d_in[p++];
      bp.qb = (const float*)d_in[p++];    bp.qw = (const float*)d_in[p++];
      bp.rpb = (const float*)d_in[p++];
    }
    dn_b[L] = (const float*)d_in[p++];
    dn_g[L] = (const float*)d_in[p++];
    red[L]  = (const float*)d_in[p++];
  }
  const float* patch_b = (const float*)d_in[p++];
  const float* patch_w = (const float*)d_in[p++];

  // scratch layout (byte offsets in d_ws)
  char* ws = (char*)d_ws;
  float* X   = (float*)(ws + ((size_t)  0 << 20));  // features (b,d,h,w,c)
  float* X2  = (float*)(ws + ((size_t)  8 << 20));  // post-attn residual
  float* XN  = (float*)(ws + ((size_t) 16 << 20));  // layernorm output
  float* XW  = (float*)(ws + ((size_t) 24 << 20));  // windowed tokens
  float* QKV = (float*)(ws + ((size_t) 36 << 20));  // qkv projections
  float* ATT = (float*)(ws + ((size_t) 64 << 20));  // attention output
  float* PRJ = (float*)(ws + ((size_t) 74 << 20));  // proj output
  float* HID = (float*)(ws + ((size_t) 84 << 20));  // MLP hidden
  float* Y   = (float*)(ws + ((size_t)108 << 20));  // MLP output
  float* MRG = (float*)(ws + ((size_t)116 << 20));  // patch-merge gather

  // ---- patch embedding: (2,1,96^3) -> (2,24,24,24,48) ----
  {
    size_t total = (size_t)B * 24 * 24 * 24 * 48;
    patch_embed_kernel<<<cdiv(total, 256), 256, 0, stream>>>(
        xin, patch_w, patch_b, X, B, 96, 24, 48);
  }

  int Dd = 24;   // cubic spatial extent
  int dim = 48;

  for (int L = 0; L < 4; ++L) {
    const int nh = NUM_HEADS_H[L];
    const int wsz = (Dd <= 7) ? Dd : 7;
    const int dp = ((Dd + wsz - 1) / wsz) * wsz;
    const int nwa = dp / wsz;
    const int nwin = nwa * nwa * nwa;
    const int b_ = B * nwin;
    const int n = wsz * wsz * wsz;
    const int tokens = B * Dd * Dd * Dd;
    const size_t tc = (size_t)tokens * dim;

    for (int j = 0; j < 2; ++j) {
      const BlockP& bp = blk[L][j];
      const int shifted = (j == 1 && wsz == 7) ? 1 : 0;
      const int ssv = shifted ? 3 : 0;
      const int M = b_ * n;

      // LN1
      layernorm_kernel<<<tokens, WAVE, 0, stream>>>(X, bp.n1g, bp.n1b, XN, tokens, dim);
      // pad + roll + window partition
      {
        size_t total = (size_t)M * dim;
        win_gather_kernel<<<cdiv(total, 256), 256, 0, stream>>>(
            XN, XW, Dd, Dd, Dd, dim, wsz, wsz, wsz, nwa, nwa, nwa,
            dp, dp, dp, ssv, total);
      }
      // QKV projection
      {
        dim3 g(cdiv(3 * dim, 32), cdiv(M, 32));
        gemm_wmma_kernel<<<g, 128, 0, stream>>>(XW, bp.qw, bp.qb, QKV,
                                                M, 3 * dim, dim, 0);
      }
      // flash attention per (window, head, q-block)
      {
        dim3 g(cdiv(n, 16), nh, b_);
        attn_flash_kernel<<<g, WAVE, 0, stream>>>(
            QKV, ATT, bp.rpb, n, dim, nh, wsz, wsz, wsz, nwa, nwa, nwa,
            dp, dp, dp, ssv, shifted);
      }
      // output projection
      {
        dim3 g(cdiv(dim, 32), cdiv(M, 32));
        gemm_wmma_kernel<<<g, 128, 0, stream>>>(ATT, bp.pw, bp.pb, PRJ,
                                                M, dim, dim, 0);
      }
      // window reverse + roll back + crop + residual
      {
        win_scatter_res_kernel<<<cdiv(tc, 256), 256, 0, stream>>>(
            X, PRJ, X2, Dd, Dd, Dd, dim, wsz, wsz, wsz, nwa, nwa, nwa,
            dp, dp, dp, ssv, tc);
      }
      // LN2 + MLP
      layernorm_kernel<<<tokens, WAVE, 0, stream>>>(X2, bp.n2g, bp.n2b, XN, tokens, dim);
      {
        dim3 g(cdiv(4 * dim, 32), cdiv(tokens, 32));
        gemm_wmma_kernel<<<g, 128, 0, stream>>>(XN, bp.fc1_w, bp.fc1_b, HID,
                                                tokens, 4 * dim, dim, 1);
      }
      {
        dim3 g(cdiv(dim, 32), cdiv(tokens, 32));
        gemm_wmma_kernel<<<g, 128, 0, stream>>>(HID, bp.fc2_w, bp.fc2_b, Y,
                                                tokens, dim, 4 * dim, 0);
      }
      add_kernel<<<cdiv(tc, 256), 256, 0, stream>>>(X2, Y, X, tc);
    }

    // ---- patch merging: (Dd^3, dim) -> (D2^3, 2*dim) ----
    const int D2 = (Dd + 1) / 2;
    const int mrows = B * D2 * D2 * D2;
    {
      size_t total = (size_t)mrows * 8 * dim;
      merge_gather_kernel<<<cdiv(total, 256), 256, 0, stream>>>(
          X, MRG, Dd, Dd, Dd, dim, D2, D2, D2, total);
    }
    layernorm_kernel<<<mrows, WAVE, 0, stream>>>(MRG, dn_g[L], dn_b[L], XN, mrows, 8 * dim);
    {
      dim3 g(cdiv(2 * dim, 32), cdiv(mrows, 32));
      gemm_wmma_kernel<<<g, 128, 0, stream>>>(XN, red[L], nullptr, X,
                                              mrows, 2 * dim, 8 * dim, 0);
    }
    Dd = D2;
    dim *= 2;
  }

  // ---- final LN (no affine) + classifier: (B*8, 768) @ (3,768)^T ----
  {
    const int rows = B * Dd * Dd * Dd;  // 2 * 2^3 = 16
    layernorm_kernel<<<rows, WAVE, 0, stream>>>(X, nullptr, nullptr, XN, rows, dim);
    dim3 g(cdiv(3, 32), cdiv(rows, 32));
    gemm_wmma_kernel<<<g, 128, 0, stream>>>(XN, cls_w, cls_b, (float*)d_out,
                                            rows, 3, dim, 0);
  }
}
